// PhysicsAttention_9663676416281
// MI455X (gfx1250) — matrix operand order repaired and checked by reading, not compiled, not run on hardware
//
#include <hip/hip_runtime.h>
#include <math.h>

// Problem constants (reference: B,N,D = 4,16384,256; M,H = 64,8; HD = 32)
#define BB 4
#define NN 16384
#define DD 256
#define MM 64
#define HH 8
#define HDD 32
#define NSPLIT 32

typedef __bf16 bf16_t;
typedef __attribute__((ext_vector_type(16))) __bf16 v16bf;
typedef __attribute__((ext_vector_type(8)))  __bf16 bf16x8;
typedef __attribute__((ext_vector_type(8)))  float  v8f;
typedef __attribute__((ext_vector_type(4)))  float  f32x4;

union V16U { v16bf v; bf16x8 h[2]; };

// ---------------------------------------------------------------------------
// K0a: fp32 -> bf16 grid-stride convert
// ---------------------------------------------------------------------------
__global__ void k_convert(const float* __restrict__ src, bf16_t* __restrict__ dst, int n) {
  int i = blockIdx.x * blockDim.x + threadIdx.x;
  int stride = gridDim.x * blockDim.x;
  for (; i < n; i += stride) dst[i] = (bf16_t)src[i];
}

// ---------------------------------------------------------------------------
// K0b: x [B][N][D] fp32 -> xT [B][D][N] bf16, tiled transpose through LDS
// grid (N/32, D/32, B), block (32,8)
// ---------------------------------------------------------------------------
__global__ void k_transpose_bf(const float* __restrict__ x, bf16_t* __restrict__ xT) {
  __shared__ float t[32][33];
  const int b = blockIdx.z;
  const int n0 = blockIdx.x * 32, d0 = blockIdx.y * 32;
  const float* xs = x + (size_t)b * NN * DD;
  for (int r = threadIdx.y; r < 32; r += 8)
    t[r][threadIdx.x] = xs[(size_t)(n0 + r) * DD + d0 + threadIdx.x];
  __syncthreads();
  bf16_t* xd = xT + (size_t)b * DD * NN;
  for (int r = threadIdx.y; r < 32; r += 8)
    xd[(size_t)(d0 + r) * NN + n0 + threadIdx.x] = (bf16_t)t[threadIdx.x][r];
}

// ---------------------------------------------------------------------------
// K1: logits[b,n,m] = x·Ws^T + bs via v_wmma_f32_16x16x32_bf16.
// Block = 256 threads = 8 waves; each wave owns 16 n-rows, all 64 m.
// Results staged in LDS, then stored coalesced in BOTH [B][M][N] and [B][N][M].
// grid (N/128, B)
// ---------------------------------------------------------------------------
__global__ void __launch_bounds__(256) k_logits(
    const bf16_t* __restrict__ xb, const bf16_t* __restrict__ wsb,
    const float* __restrict__ bs,
    float* __restrict__ Lmn, float* __restrict__ Lnm) {
  __shared__ float tile[128][65];   // padded: avoid 32-way bank conflicts on transpose read
  const int b   = blockIdx.y;
  const int n0  = blockIdx.x * 128;
  const int tid = threadIdx.x;
  const int lane = tid & 31, wid = tid >> 5;
  const int lo = lane & 15, hi = lane >> 4;
  const int nrow = n0 + wid * 16 + lo;
  const bf16_t* xrow = xb + ((size_t)b * NN + nrow) * DD;

  v8f c0{}, c1{}, c2{}, c3{};
  for (int kd = 0; kd < DD; kd += 32) {
    __builtin_prefetch(xrow + kd + 64, 0, 1);    // global_prefetch_b8 next chunk
    V16U a;
    a.h[0] = *(const bf16x8*)(xrow + kd + 16 * hi);        // K = kd+16hi .. +7
    a.h[1] = *(const bf16x8*)(xrow + kd + 16 * hi + 8);    // K = kd+16hi+8 .. +7
    v16bf b0 = *(const v16bf*)(wsb + (size_t)(0 * 16 + lo) * DD + kd + 16 * hi);
    v16bf b1 = *(const v16bf*)(wsb + (size_t)(1 * 16 + lo) * DD + kd + 16 * hi);
    v16bf b2 = *(const v16bf*)(wsb + (size_t)(2 * 16 + lo) * DD + kd + 16 * hi);
    v16bf b3 = *(const v16bf*)(wsb + (size_t)(3 * 16 + lo) * DD + kd + 16 * hi);
    c0 = __builtin_amdgcn_wmma_f32_16x16x32_bf16(false, a.v, false, b0, (short)0, c0, false, false);
    c1 = __builtin_amdgcn_wmma_f32_16x16x32_bf16(false, a.v, false, b1, (short)0, c1, false, false);
    c2 = __builtin_amdgcn_wmma_f32_16x16x32_bf16(false, a.v, false, b2, (short)0, c2, false, false);
    c3 = __builtin_amdgcn_wmma_f32_16x16x32_bf16(false, a.v, false, b3, (short)0, c3, false, false);
  }
  // C layout: VGPR r -> row r (lanes 0-15) / r+8 (lanes 16-31), col = lane%16
#pragma unroll
  for (int r = 0; r < 8; ++r) {
    const int nl = wid * 16 + r + 8 * hi;
    tile[nl][ 0 + lo] = c0[r] + bs[ 0 + lo];
    tile[nl][16 + lo] = c1[r] + bs[16 + lo];
    tile[nl][32 + lo] = c2[r] + bs[32 + lo];
    tile[nl][48 + lo] = c3[r] + bs[48 + lo];
  }
  __syncthreads();
  for (int i = tid; i < MM * 128; i += 256) {          // [B][M][N], coalesced in n
    const int m = i >> 7, nl = i & 127;
    Lmn[((size_t)b * MM + m) * NN + n0 + nl] = tile[nl][m];
  }
  for (int i = tid; i < 128 * MM; i += 256) {          // [B][N][M], coalesced in m
    const int nl = i >> 6, m = i & 63;
    Lnm[((size_t)b * NN + n0 + nl) * MM + m] = tile[nl][m];
  }
}

// ---------------------------------------------------------------------------
// K2: per-(b,m) softmax stats over N=16384: {max, 1/sum(exp)}  — grid B*M
// ---------------------------------------------------------------------------
__global__ void __launch_bounds__(256) k_stats(const float* __restrict__ Lmn,
                                               float2* __restrict__ stats) {
  const int bm = blockIdx.x;
  const float* row = Lmn + (size_t)bm * NN;
  __shared__ float red[256];
  const int tid = threadIdx.x;
  float mx = -1e30f;
  for (int i = tid; i < NN; i += 256) mx = fmaxf(mx, row[i]);
  red[tid] = mx; __syncthreads();
  for (int s = 128; s > 0; s >>= 1) { if (tid < s) red[tid] = fmaxf(red[tid], red[tid + s]); __syncthreads(); }
  mx = red[0]; __syncthreads();
  float sum = 0.f;
  for (int i = tid; i < NN; i += 256) sum += __expf(row[i] - mx);
  red[tid] = sum; __syncthreads();
  for (int s = 128; s > 0; s >>= 1) { if (tid < s) red[tid] += red[tid + s]; __syncthreads(); }
  if (tid == 0) stats[bm] = make_float2(mx, 1.0f / red[0]);
}

// ---------------------------------------------------------------------------
// K3: tokens[b,m,d] = sum_n softmax(L)[m,n] * x[n,d]  (WMMA, K=N split over
// NSPLIT blocks into partial buffers). A = w (built on the fly, bf16),
// B = xT[d][n] (contiguous-K 32B lane loads).  grid (NSPLIT, B)
// ---------------------------------------------------------------------------
__global__ void __launch_bounds__(256) k_tokens(
    const float* __restrict__ Lmn, const float2* __restrict__ stats,
    const bf16_t* __restrict__ xbT, float* __restrict__ part) {
  const int b = blockIdx.y, split = blockIdx.x;
  const int tid = threadIdx.x;
  const int lane = tid & 31, wid = tid >> 5;
  const int lo = lane & 15, hi = lane >> 4;
  const int mt = wid & 3;      // m tile: rows mt*16..+15
  const int dg = wid >> 2;     // d group: 0 -> d 0..127, 1 -> d 128..255
  const int m  = mt * 16 + lo; // this lane's A-matrix row
  const float2 st = stats[b * MM + m];
  const float mx = st.x, inv = st.y;
  const float*  lrow = Lmn + ((size_t)b * MM + m) * NN;
  const bf16_t* xT   = xbT + (size_t)b * DD * NN;

  v8f acc[8];
#pragma unroll
  for (int i = 0; i < 8; ++i) acc[i] = v8f{};

  const int nbeg = split * (NN / NSPLIT), nend = nbeg + NN / NSPLIT;
  for (int n = nbeg; n < nend; n += 32) {
    __builtin_prefetch(lrow + n + 64, 0, 1);
    V16U a;
    const float* p0 = lrow + n + 16 * hi;
    const float* p1 = lrow + n + 16 * hi + 8;
#pragma unroll
    for (int j = 0; j < 8; ++j) {
      a.h[0][j] = (bf16_t)(__expf(p0[j] - mx) * inv);
      a.h[1][j] = (bf16_t)(__expf(p1[j] - mx) * inv);
    }
#pragma unroll
    for (int dt = 0; dt < 8; ++dt) {
      const int d = dg * 128 + dt * 16 + lo;
      v16bf bf = *(const v16bf*)(xT + (size_t)d * NN + n + 16 * hi);
      acc[dt] = __builtin_amdgcn_wmma_f32_16x16x32_bf16(false, a.v, false, bf, (short)0, acc[dt], false, false);
    }
  }
  float* pout = part + ((size_t)split * BB + b) * MM * DD;
#pragma unroll
  for (int dt = 0; dt < 8; ++dt)
#pragma unroll
    for (int r = 0; r < 8; ++r) {
      const int mrow = mt * 16 + r + 8 * hi;
      const int d = dg * 128 + dt * 16 + lo;
      pout[mrow * DD + d] = acc[dt][r];
    }
}

__global__ void k_tokens_reduce(const float* __restrict__ part, float* __restrict__ tokens) {
  const int i = blockIdx.x * 256 + threadIdx.x;   // B*M*D = 65536
  if (i >= BB * MM * DD) return;
  const int b = i / (MM * DD), rem = i % (MM * DD);
  float s = 0.f;
  for (int sp = 0; sp < NSPLIT; ++sp) s += part[((size_t)sp * BB + b) * MM * DD + rem];
  tokens[i] = s;
}

// ---------------------------------------------------------------------------
// K4a: qkv[b,m,e] = tokens[b,m,:] . Wqkv[e,:]   (tiny fp32 GEMM) grid B*M
// ---------------------------------------------------------------------------
__global__ void __launch_bounds__(256) k_qkv(const float* __restrict__ tokens,
                                             const float* __restrict__ Wqkv,
                                             float* __restrict__ qkv) {
  const int bm = blockIdx.x;
  __shared__ float t[DD];
  t[threadIdx.x] = tokens[(size_t)bm * DD + threadIdx.x];
  __syncthreads();
  for (int e = threadIdx.x; e < 3 * DD; e += 256) {
    const float* w = Wqkv + (size_t)e * DD;
    float s = 0.f;
    for (int d = 0; d < DD; ++d) s += t[d] * w[d];
    qkv[(size_t)bm * 3 * DD + e] = s;
  }
}

// ---------------------------------------------------------------------------
// K4b: 64-token attention per (b,h). grid B*H, block 64 (thread = query row)
// ---------------------------------------------------------------------------
__global__ void __launch_bounds__(64) k_attn(const float* __restrict__ qkv,
                                             float* __restrict__ t2) {
  const int b = blockIdx.x / HH, h = blockIdx.x % HH;
  const int m = threadIdx.x;
  __shared__ float Ksh[MM][HDD], Vsh[MM][HDD];
  const float* base = qkv + (size_t)b * MM * 3 * DD;
  for (int i = threadIdx.x; i < MM * HDD; i += 64) {
    const int mm = i / HDD, d = i % HDD;
    Ksh[mm][d] = base[(size_t)mm * 3 * DD + DD     + h * HDD + d];
    Vsh[mm][d] = base[(size_t)mm * 3 * DD + 2 * DD + h * HDD + d];
  }
  __syncthreads();
  float q[HDD];
  for (int d = 0; d < HDD; ++d) q[d] = base[(size_t)m * 3 * DD + h * HDD + d];
  const float scale = 0.17677669529663687f;   // 1/sqrt(32)
  float s[MM]; float mx = -1e30f;
  for (int k = 0; k < MM; ++k) {
    float acc = 0.f;
    for (int d = 0; d < HDD; ++d) acc += q[d] * Ksh[k][d];
    s[k] = acc * scale; mx = fmaxf(mx, s[k]);
  }
  float den = 0.f;
  for (int k = 0; k < MM; ++k) { s[k] = __expf(s[k] - mx); den += s[k]; }
  const float invd = 1.f / den;
  float o[HDD];
  for (int d = 0; d < HDD; ++d) o[d] = 0.f;
  for (int k = 0; k < MM; ++k) {
    const float w = s[k] * invd;
    for (int d = 0; d < HDD; ++d) o[d] += w * Vsh[k][d];
  }
  float* out = t2 + ((size_t)b * MM + m) * DD + h * HDD;
  for (int d = 0; d < HDD; ++d) out[d] = o[d];
}

// ---------------------------------------------------------------------------
// K4c: token_out = t2·Wo^T + bo, stored TRANSPOSED in bf16: toT[b][d][m]
// ---------------------------------------------------------------------------
__global__ void __launch_bounds__(256) k_oproj(const float* __restrict__ t2,
                                               const float* __restrict__ Wo,
                                               const float* __restrict__ bo,
                                               bf16_t* __restrict__ toT) {
  const int bm = blockIdx.x, b = bm >> 6, m = bm & 63;
  __shared__ float t[DD];
  t[threadIdx.x] = t2[(size_t)bm * DD + threadIdx.x];
  __syncthreads();
  for (int e = threadIdx.x; e < DD; e += 256) {
    const float* w = Wo + (size_t)e * DD;
    float s = bo[e];
    for (int d = 0; d < DD; ++d) s += t[d] * w[d];
    toT[((size_t)b * DD + e) * MM + m] = (bf16_t)s;
  }
}

// ---------------------------------------------------------------------------
// K5: out[b,n,d] = sum_m w[n,m] * token_out[m,d]   (WMMA, K = M = 64)
// A built on the fly from Lnm (contiguous in m) + stats in LDS.
// B = toT[d][m] (contiguous-K 32B lane loads). grid (N/128, 2, B)
// ---------------------------------------------------------------------------
__global__ void __launch_bounds__(256) k_unpool(
    const float* __restrict__ Lnm, const float2* __restrict__ stats,
    const bf16_t* __restrict__ toT, float* __restrict__ out) {
  const int b = blockIdx.z;
  const int dhalf = blockIdx.y;
  const int n0 = blockIdx.x * 128;
  const int tid = threadIdx.x;
  const int lane = tid & 31, wid = tid >> 5;
  const int lo = lane & 15, hi = lane >> 4;
  __shared__ float smx[MM], sinv[MM];
  if (tid < MM) { float2 st = stats[b * MM + tid]; smx[tid] = st.x; sinv[tid] = st.y; }
  __syncthreads();
  const int nrow = n0 + wid * 16 + lo;
  const float*  lrow = Lnm + ((size_t)b * NN + nrow) * MM;
  const bf16_t* tT   = toT + (size_t)b * DD * MM;

  v8f acc[8];
#pragma unroll
  for (int i = 0; i < 8; ++i) acc[i] = v8f{};

#pragma unroll
  for (int k0 = 0; k0 < MM; k0 += 32) {
    V16U a;
#pragma unroll
    for (int j = 0; j < 8; ++j) {
      const int m0 = k0 + 16 * hi + j;
      const int m1 = k0 + 16 * hi + 8 + j;
      a.h[0][j] = (bf16_t)(__expf(lrow[m0] - smx[m0]) * sinv[m0]);
      a.h[1][j] = (bf16_t)(__expf(lrow[m1] - smx[m1]) * sinv[m1]);
    }
#pragma unroll
    for (int dt = 0; dt < 8; ++dt) {
      const int d = dhalf * 128 + dt * 16 + lo;
      v16bf bf = *(const v16bf*)(tT + (size_t)d * MM + k0 + 16 * hi);
      acc[dt] = __builtin_amdgcn_wmma_f32_16x16x32_bf16(false, a.v, false, bf, (short)0, acc[dt], false, false);
    }
  }
  float* ob = out + (size_t)b * NN * DD;
#pragma unroll
  for (int dt = 0; dt < 8; ++dt)
#pragma unroll
    for (int r = 0; r < 8; ++r) {
      const int n = n0 + wid * 16 + r + 8 * hi;
      const int d = dhalf * 128 + dt * 16 + lo;
      ob[(size_t)n * DD + d] = acc[dt][r];    // coalesced in d
    }
}

// ---------------------------------------------------------------------------
extern "C" void kernel_launch(void* const* d_in, const int* in_sizes, int n_in,
                              void* d_out, int out_size, void* d_ws, size_t ws_size,
                              hipStream_t stream) {
  const float* x    = (const float*)d_in[0];   // [B,N,D]
  const float* Ws   = (const float*)d_in[1];   // [M,D]
  const float* bs   = (const float*)d_in[2];   // [M]
  const float* Wqkv = (const float*)d_in[3];   // [3D,D]
  const float* Wo   = (const float*)d_in[4];   // [D,D]
  const float* bo   = (const float*)d_in[5];   // [D]
  float* out = (float*)d_out;                  // [B,N,D]

  char* ws = (char*)d_ws;
  size_t off = 0;
  auto alloc = [&](size_t bytes) { char* p = ws + off; off = (off + bytes + 255) & ~(size_t)255; return p; };

  bf16_t* xb     = (bf16_t*)alloc((size_t)BB * NN * DD * 2);          // x bf16 [B][N][D]
  bf16_t* xbT    = (bf16_t*)alloc((size_t)BB * DD * NN * 2);          // x bf16 [B][D][N]
  bf16_t* wsb    = (bf16_t*)alloc((size_t)MM * DD * 2);               // Ws bf16
  float*  Lmn    = (float*) alloc((size_t)BB * MM * NN * 4);          // logits [B][M][N]
  float*  Lnm    = (float*) alloc((size_t)BB * NN * MM * 4);          // logits [B][N][M]
  float2* stats  = (float2*)alloc((size_t)BB * MM * 8);               // {max, 1/den}
  float*  part   = (float*) alloc((size_t)NSPLIT * BB * MM * DD * 4); // token partials
  float*  tokens = (float*) alloc((size_t)BB * MM * DD * 4);
  float*  qkv    = (float*) alloc((size_t)BB * MM * 3 * DD * 4);
  float*  t2     = (float*) alloc((size_t)BB * MM * DD * 4);          // attn out
  bf16_t* toT    = (bf16_t*)alloc((size_t)BB * DD * MM * 2);          // token_out^T bf16

  k_convert<<<4096, 256, 0, stream>>>(x, xb, BB * NN * DD);
  k_convert<<<64, 256, 0, stream>>>(Ws, wsb, MM * DD);
  k_transpose_bf<<<dim3(NN / 32, DD / 32, BB), dim3(32, 8), 0, stream>>>(x, xbT);

  k_logits<<<dim3(NN / 128, BB), 256, 0, stream>>>(xb, wsb, bs, Lmn, Lnm);
  k_stats<<<BB * MM, 256, 0, stream>>>(Lmn, stats);
  k_tokens<<<dim3(NSPLIT, BB), 256, 0, stream>>>(Lmn, stats, xbT, part);
  k_tokens_reduce<<<(BB * MM * DD + 255) / 256, 256, 0, stream>>>(part, tokens);

  k_qkv<<<BB * MM, 256, 0, stream>>>(tokens, Wqkv, qkv);
  k_attn<<<BB * HH, 64, 0, stream>>>(qkv, t2);
  k_oproj<<<BB * MM, 256, 0, stream>>>(t2, Wo, bo, toT);

  k_unpool<<<dim3(NN / 128, 2, BB), 256, 0, stream>>>(Lnm, stats, toT, out);
}
